// TBNN_Q_direction_65644280152315
// MI455X (gfx1250) — matrix operand order repaired and checked by reading, not compiled, not run on hardware
//
#include <hip/hip_runtime.h>

typedef __attribute__((ext_vector_type(16))) _Float16 v16h;
typedef __attribute__((ext_vector_type(8)))  _Float16 v8h;
typedef __attribute__((ext_vector_type(8)))  float    v8f;
typedef unsigned int u32x4 __attribute__((ext_vector_type(4)));
typedef int          i32x8 __attribute__((ext_vector_type(8)));
typedef int          i32x4 __attribute__((ext_vector_type(4)));

#define BLOCK 128

namespace {
// layer dims (in K, out N), padded K to 32, N to 16
constexpr int Kd[6] = {5, 50, 100, 100, 100, 50};
constexpr int Nd[6] = {50, 100, 100, 100, 50, 10};
constexpr int KpA[6] = {32, 64, 128, 128, 128, 64};
constexpr int NpA[6] = {64, 128, 128, 128, 64, 16};
// half-element offsets of each layer's packed weights
constexpr int WB[6] = {0, 2048, 10240, 26624, 43008, 51200};
// float-element offsets of each layer's padded bias
constexpr int BB[6] = {0, 64, 192, 320, 448, 512};
constexpr int WTOT = 52224;                       // halfs
constexpr int BTOT = 528;                         // floats
constexpr int PARAM_BYTES = WTOT * 2 + BTOT * 4;  // 106560 B, contiguous blob
constexpr int PARAM_QW = PARAM_BYTES / 8;         // 13320 8-byte elements
}

// ---------- small 3x3 helpers (row-major float[9]) ----------
__device__ __forceinline__ void mm3(const float* a, const float* b, float* c) {
#pragma unroll
  for (int i = 0; i < 3; ++i)
#pragma unroll
    for (int j = 0; j < 3; ++j)
      c[i * 3 + j] = a[i * 3 + 0] * b[0 * 3 + j] + a[i * 3 + 1] * b[1 * 3 + j] +
                     a[i * 3 + 2] * b[2 * 3 + j];
}
__device__ __forceinline__ float tr3(const float* a) { return a[0] + a[4] + a[8]; }
// trace(a @ b) without forming the product
__device__ __forceinline__ float trAB(const float* a, const float* b) {
  float r = 0.0f;
#pragma unroll
  for (int i = 0; i < 3; ++i)
#pragma unroll
    for (int j = 0; j < 3; ++j) r += a[i * 3 + j] * b[j * 3 + i];
  return r;
}

// ---------- kernel 1: prepack weights into WMMA B-operand bricks ----------
// Brick = one 32(K) x 16(N) sub-block, stored as 32 lanes x 16 halfs:
//   lane L (<16):  col N=L,    holds K = kstep*32 + 0..15   (half index h)
//   lane L (>=16): col N=L-16, holds K = kstep*32 + 16..31
struct PackParams {
  const float* W[6];
  const float* Bv[6];
  _Float16* wout;
  float* bout;
};

__global__ void tbnn_prepack_kernel(PackParams p) {
  const int tid = blockIdx.x * blockDim.x + threadIdx.x;
  const int stride = gridDim.x * blockDim.x;
#pragma unroll
  for (int l = 0; l < 6; ++l) {
    const int K = Kd[l], N = Nd[l];
    const int NT = NpA[l] >> 4;         // N tiles
    const int total = KpA[l] * NpA[l];  // padded halfs this layer
    const float* Wsrc = p.W[l];
    for (int i = tid; i < total; i += stride) {
      const int brick = i >> 9;
      const int rem = i & 511;
      const int L = rem >> 4;
      const int h = rem & 15;
      const int kstep = brick / NT;
      const int ntile = brick - kstep * NT;
      const int k = kstep * 32 + ((L >> 4) << 4) + h;
      const int n = ntile * 16 + (L & 15);
      const float v = (k < K && n < N) ? Wsrc[k * N + n] : 0.0f;
      p.wout[WB[l] + i] = (_Float16)v;
    }
    const float* Bsrc = p.Bv[l];
    for (int i = tid; i < NpA[l]; i += stride)
      p.bout[BB[l] + i] = (i < N) ? Bsrc[i] : 0.0f;
  }
}

// ---------- one MLP layer on BOTH 16-row tiles of a wave via WMMA ----------
// Shares each B brick across two independent accumulator streams: halves
// B-operand LDS reads and provides independent WMMAs to cover the
// WMMA->WMMA data hazard slots.
template <int KP, int NP, bool LAST>
__device__ __forceinline__ void mlp_layer2(_Float16* act, const _Float16* wl,
                                           const float* bl, int rb, int lane) {
  constexpr int NT = NP / 16;
  constexpr int KS = KP / 32;
  v8f acc0[NT], acc1[NT];
#pragma unroll
  for (int n = 0; n < NT; ++n)
#pragma unroll
    for (int j = 0; j < 8; ++j) { acc0[n][j] = 0.0f; acc1[n][j] = 0.0f; }

  const int hsel = (lane >> 4) * 16;  // A layout K-half select
  const _Float16* arow0 = act + (rb + (lane & 15)) * 128;
  const _Float16* arow1 = arow0 + 16 * 128;
#pragma unroll
  for (int ks = 0; ks < KS; ++ks) {
    v16h a0, a1;
    {
      const v8h lo0 = *(const v8h*)(arow0 + ks * 32 + hsel);
      const v8h hi0 = *(const v8h*)(arow0 + ks * 32 + 8 + hsel);
      const v8h lo1 = *(const v8h*)(arow1 + ks * 32 + hsel);
      const v8h hi1 = *(const v8h*)(arow1 + ks * 32 + 8 + hsel);
#pragma unroll
      for (int i = 0; i < 8; ++i) {
        a0[i] = lo0[i]; a0[i + 8] = hi0[i];
        a1[i] = lo1[i]; a1[i + 8] = hi1[i];
      }
    }
#pragma unroll
    for (int n = 0; n < NT; ++n) {
      const v16h b = *(const v16h*)(wl + (ks * NT + n) * 512 + lane * 16);
      acc0[n] = __builtin_amdgcn_wmma_f32_16x16x32_f16(
          false, a0, false, b, (short)0, acc0[n], false, false);
      acc1[n] = __builtin_amdgcn_wmma_f32_16x16x32_f16(
          false, a1, false, b, (short)0, acc1[n], false, false);
    }
  }
  // D layout: VGPR j -> row rb + j + 8*(lane>=16), col = (lane&15) + 16*n
  const int orow = rb + ((lane >> 4) << 3);
  const int col = lane & 15;
#pragma unroll
  for (int n = 0; n < NT; ++n) {
    const float bias = bl[n * 16 + col];
#pragma unroll
    for (int j = 0; j < 8; ++j) {
      const float v0 = acc0[n][j] + bias;
      const float v1 = acc1[n][j] + bias;
      if (LAST) {
        // final g written as f32 into the free upper half of the row
        ((float*)(act + (orow + j) * 128 + 64))[n * 16 + col] = v0;
        ((float*)(act + (orow + 16 + j) * 128 + 64))[n * 16 + col] = v1;
      } else {
        // leaky_relu(0.1): max(x, 0.1x) — no cmp/cndmask
        act[(orow + j) * 128 + n * 16 + col] = (_Float16)fmaxf(v0, 0.1f * v0);
        act[(orow + 16 + j) * 128 + n * 16 + col] = (_Float16)fmaxf(v1, 0.1f * v1);
      }
    }
  }
}

// ---------- kernel 2: full fused TBNN ----------
struct MainParams {
  const float* s;
  const float* w;
  const _Float16* wpack;  // contiguous blob: weights then biases
  float* out;
  int nB;
};

__global__ void __launch_bounds__(BLOCK) tbnn_main_kernel(MainParams p) {
  __shared__ __align__(16) unsigned char s_params[PARAM_BYTES];  // 106560 B
  __shared__ _Float16 s_act[BLOCK * 128];                        //  32768 B
  _Float16* s_wgt = (_Float16*)s_params;
  float* s_bias = (float*)(s_params + WTOT * 2);

  const int tid = threadIdx.x;
  const int lane = tid & 31;
  const int waveId = tid >> 5;

  // ---- stage packed params global->LDS ----
#if defined(__gfx1250__) && __has_builtin(__builtin_amdgcn_tensor_load_to_lds)
  const bool tdmWave = (waveId == 0);  // wave-uniform: TDM ignores EXEC
  if (tdmWave) {
    // Tensor DMA descriptor (ISA 8.3-8.6): 1D tile of PARAM_QW 8-byte elems
    const unsigned long long ga = (unsigned long long)p.wpack;
    const unsigned int lds = (unsigned int)(uintptr_t)(void*)s_params;
    u32x4 g0;
    g0[0] = 1u;                                 // count=1 (valid user D#)
    g0[1] = lds;                                // lds_addr
    g0[2] = (unsigned int)(ga & 0xffffffffu);   // global_addr[31:0]
    g0[3] = (unsigned int)((ga >> 32) & 0x01ffffffu) | 0x80000000u;  // [56:32] | type=2
    i32x8 g1;
    g1[0] = 3 << 16;                            // data_size=8B; wg_mask=0
    g1[1] = (int)((PARAM_QW & 0xffff) << 16);   // tensor_dim0[15:0]
    g1[2] = (int)((PARAM_QW >> 16) & 0xffff) | (1 << 16);  // dim0 hi | tensor_dim1=1
    g1[3] = (int)((PARAM_QW & 0xffff) << 16);   // tile_dim0
    g1[4] = 1;                                  // tile_dim1=1, tile_dim2=0
    g1[5] = (int)PARAM_QW;                      // tensor_dim0_stride lo32
    g1[6] = 0;
    g1[7] = 0;
    const i32x4 gz = {0, 0, 0, 0};              // groups 2/3 unused (<=2D)
#if __clang_major__ >= 23
    const i32x8 gz8 = {0, 0, 0, 0, 0, 0, 0, 0};
    __builtin_amdgcn_tensor_load_to_lds(g0, g1, gz, gz, gz8, 0);
#else
    __builtin_amdgcn_tensor_load_to_lds(g0, g1, gz, gz, 0);
#endif
  }
#else
  {
    const v8h* src = (const v8h*)p.wpack;
    v8h* dst = (v8h*)s_params;
    for (int i = tid; i < PARAM_BYTES / 16; i += BLOCK) dst[i] = src[i];
  }
#endif

  // ---- phase 1: invariants (one element per thread), overlaps the DMA ----
  const int gid = blockIdx.x * BLOCK + tid;
  const bool valid = gid < p.nB;
  float S[9], W9[9];
#pragma unroll
  for (int i = 0; i < 9; ++i) {
    S[i]  = valid ? p.s[gid * 9 + i] : 0.0f;
    W9[i] = valid ? p.w[gid * 9 + i] : 0.0f;
  }
  float s2[9], w2[9];
  mm3(S, S, s2);
  mm3(W9, W9, w2);
  const float lam1 = tr3(s2);
  const float lam2 = tr3(w2);
  const float lam3 = trAB(s2, S);
  const float lam4 = trAB(w2, S);
  const float lam5 = trAB(w2, s2);

  {
    _Float16* r = s_act + tid * 128;
    r[0] = (_Float16)lam1;
    r[1] = (_Float16)lam2;
    r[2] = (_Float16)lam3;
    r[3] = (_Float16)lam4;
    r[4] = (_Float16)lam5;
#pragma unroll
    for (int i = 5; i < 32; ++i) r[i] = (_Float16)0.0f;
  }

#if defined(__gfx1250__) && __has_builtin(__builtin_amdgcn_tensor_load_to_lds)
  if (tdmWave) __builtin_amdgcn_s_wait_tensorcnt(0);
#endif
  __syncthreads();

  // ---- phase 2: MLP, each wave drives its two 16-row tiles per layer ----
  {
    const int rb = waveId * 32;
    mlp_layer2<32, 64, false>(s_act, s_wgt + WB[0], s_bias + BB[0], rb, lane);
    __syncthreads();
    mlp_layer2<64, 128, false>(s_act, s_wgt + WB[1], s_bias + BB[1], rb, lane);
    __syncthreads();
    mlp_layer2<128, 128, false>(s_act, s_wgt + WB[2], s_bias + BB[2], rb, lane);
    __syncthreads();
    mlp_layer2<128, 128, false>(s_act, s_wgt + WB[3], s_bias + BB[3], rb, lane);
    __syncthreads();
    mlp_layer2<128, 64, false>(s_act, s_wgt + WB[4], s_bias + BB[4], rb, lane);
    __syncthreads();
    mlp_layer2<64, 16, true>(s_act, s_wgt + WB[5], s_bias + BB[5], rb, lane);
    __syncthreads();
  }

  // ---- phase 3: tensor bases, combine, deviatoric+symmetrize+normalize ----
  float gg[10];
  {
    const float* g = (const float*)(s_act + tid * 128 + 64);
#pragma unroll
    for (int n = 0; n < 10; ++n) gg[n] = g[n];
  }

  float sw[9], ws[9], s2w[9], ws2[9], w2s[9], sw2[9], s2w2[9], w2s2[9];
  mm3(S, W9, sw);
  mm3(W9, S, ws);
  mm3(s2, W9, s2w);
  mm3(W9, s2, ws2);
  mm3(w2, S, w2s);
  mm3(S, w2, sw2);
  mm3(s2, w2, s2w2);
  mm3(w2, s2, w2s2);
  float t7a[9], t7b[9], t8a[9], t8b[9], t10a[9], t10b[9];
  mm3(ws, w2, t7a);
  mm3(w2, sw, t7b);
  mm3(sw, s2, t8a);
  mm3(s2, ws, t8b);
  mm3(ws2, w2, t10a);
  mm3(w2, s2w, t10b);
  const float trsw2 = tr3(sw2);
  const float trs2w2 = tr3(s2w2);

  float Q[9];
#pragma unroll
  for (int i = 0; i < 9; ++i) {
    const float dg = (i == 0 || i == 4 || i == 8) ? 1.0f : 0.0f;
    const float T1 = S[i];
    const float T2 = sw[i] - ws[i];
    const float T3 = s2[i] - (lam1 * (1.0f / 3.0f)) * dg;
    const float T4 = w2[i] - (lam2 * (1.0f / 3.0f)) * dg;
    const float T5 = ws2[i] - s2w[i];
    const float T6 = w2s[i] + sw2[i] - (2.0f / 3.0f) * trsw2 * dg;
    const float T7 = t7a[i] - t7b[i];
    const float T8 = t8a[i] - t8b[i];
    const float T9 = w2s2[i] + s2w2[i] - (2.0f / 3.0f) * trs2w2 * dg;
    const float T10 = t10a[i] - t10b[i];
    Q[i] = gg[0] * T1 + gg[1] * T2 + gg[2] * T3 + gg[3] * T4 + gg[4] * T5 +
           gg[5] * T6 + gg[6] * T7 + gg[7] * T8 + gg[8] * T9 + gg[9] * T10;
  }

  if (valid) {
    float raw[9];
#pragma unroll
    for (int i = 0; i < 9; ++i) raw[i] = Q[i];
    const float trQ3 = (Q[0] + Q[4] + Q[8]) * (1.0f / 3.0f);
    Q[0] -= trQ3; Q[4] -= trQ3; Q[8] -= trQ3;
    float Qs[9];
#pragma unroll
    for (int i = 0; i < 3; ++i)
#pragma unroll
      for (int j = 0; j < 3; ++j)
        Qs[i * 3 + j] = 0.5f * (Q[i * 3 + j] + Q[j * 3 + i]);
    float nsq = 0.0f;
#pragma unroll
    for (int i = 0; i < 9; ++i) nsq += Qs[i] * Qs[i];
    const float inv = rsqrtf(nsq + 1e-16f);
    float* outQ = p.out + (size_t)gid * 9;
    float* outR = p.out + (size_t)p.nB * 9 + (size_t)gid * 9;
#pragma unroll
    for (int i = 0; i < 9; ++i) outQ[i] = Qs[i] * inv;
#pragma unroll
    for (int i = 0; i < 9; ++i) outR[i] = raw[i];
  }
}

extern "C" void kernel_launch(void* const* d_in, const int* in_sizes, int n_in,
                              void* d_out, int out_size, void* d_ws, size_t ws_size,
                              hipStream_t stream) {
  (void)n_in; (void)out_size; (void)ws_size;

  PackParams pp;
  for (int l = 0; l < 6; ++l) {
    pp.W[l]  = (const float*)d_in[2 + 2 * l];
    pp.Bv[l] = (const float*)d_in[3 + 2 * l];
  }
  pp.wout = (_Float16*)d_ws;
  pp.bout = (float*)((char*)d_ws + (size_t)WTOT * sizeof(_Float16));
  tbnn_prepack_kernel<<<64, 256, 0, stream>>>(pp);

  MainParams mp;
  mp.s = (const float*)d_in[0];
  mp.w = (const float*)d_in[1];
  mp.wpack = (const _Float16*)d_ws;
  mp.out = (float*)d_out;
  mp.nB = in_sizes[0] / 9;
  const int grid = (mp.nB + BLOCK - 1) / BLOCK;
  tbnn_main_kernel<<<grid, BLOCK, 0, stream>>>(mp);
}
